// Block_13383118094681
// MI455X (gfx1250) — matrix operand order repaired and checked by reading, not compiled, not run on hardware
//
#include <hip/hip_runtime.h>
#include <math.h>

// ===========================================================================
// MI455X / gfx1250 implementation of the 4-layer retention + AFT/SDPA model.
//
//  * All GEMMs (~945 GFLOP) run through V_WMMA_F32_16X16X32_F16 (f16 in,
//    f32 accumulate); inputs are rms-normalized / silu-gated => f16-safe.
//  * Global->LDS staging uses GLOBAL_LOAD_ASYNC_TO_LDS_B128 (ASYNCcnt +
//    s_wait_asynccnt); int4* operands (AS1 src, AS3 dst).
//  * Single NT GEMM variant: A [m][k] row-major, B [n][k] row-major.  All
//    transposed operands (u^T, sv^T, w_out) are pre-transposed in global
//    memory by an LDS-tiled f16 transpose, so every WMMA fragment read from
//    LDS is a contiguous 16B run (ds_load_b128), never a u16 gather.
//  * Flash attention: Q/K async-staged contiguous; V transposed during
//    staging into [d][k] so P@V fragments are contiguous too.
//
// WMMA fragment layouts per CDNA5 ISA 7.12.2 (wave32):
//   A (16x32 f16): lane L row M=L%16; kbase=(L<16?0:8); elem j -> K=kbase+j+(j>=8?8:0)
//   B (32x16 f16): lane L col N=L%16; K order as for A (this code: A and B both K=16*(L/16)+j)
//   C/D (16x16 f32): lane L, reg r -> M = r + (L<16?0:8), N = L%16
// ===========================================================================

typedef _Float16 half_t;
typedef __attribute__((ext_vector_type(16))) _Float16 v16h;
typedef __attribute__((ext_vector_type(8)))  _Float16 v8h;
typedef __attribute__((ext_vector_type(8)))  float    v8f;
typedef __attribute__((ext_vector_type(4)))  int      i4;

#define RMS_EPS 1.1920929e-07f

static constexpr int Bb = 4, Tt = 1024, E = 1024, Qd = 2048, Hh = 16, Dh = 128;
static constexpr int Mr   = Bb * Tt;    // 4096 token rows
static constexpr int Pr   = 12 * E;     // 12288 "parts" rows
static constexpr int QKVW = 3 * Qd;     // 6144

#if defined(__has_builtin)
#if __has_builtin(__builtin_amdgcn_global_load_async_to_lds_b128)
#define USE_ASYNC_LDS 1
#endif
#endif

// 16-byte global -> LDS copy (async when available)
__device__ __forceinline__ void g2l_b128(const half_t* g, half_t* l) {
#ifdef USE_ASYNC_LDS
  __builtin_amdgcn_global_load_async_to_lds_b128(
      (__attribute__((address_space(1))) i4*)(unsigned long long)g,
      (__attribute__((address_space(3))) i4*)(unsigned int)(unsigned long long)l,
      0, 0);
#else
  *(v8h*)l = *(const v8h*)g;
#endif
}
__device__ __forceinline__ void async_wait() {
#ifdef USE_ASYNC_LDS
#if __has_builtin(__builtin_amdgcn_s_wait_asynccnt)
  __builtin_amdgcn_s_wait_asynccnt(0);
#else
  asm volatile("s_wait_asynccnt 0" ::: "memory");
#endif
#endif
}

__device__ __forceinline__ v8f wmma16(v16h a, v16h b, v8f c) {
  return __builtin_amdgcn_wmma_f32_16x16x32_f16(false, a, false, b, (short)0, c,
                                                false, false);
}
__device__ __forceinline__ int a_k_of(int j) { return j; }

// --------------------------------------------------------------------------
// NT WMMA GEMM:  Cout = alpha * (Cin + A @ B^T)      (Cin optional)
//   A: M x K row-major (lda), B: N x K row-major (ldb)  ->  C: M x N (ldc)
// Block: 256 threads = 8 waves -> 64(M) x 128(N); wave = 16x64; K-step 32.
// LDS tiles [row][k]: staging and fragment reads are both 16B-contiguous.
// M%64==0, N%128==0, K%32==0 by construction of the model dims.
// --------------------------------------------------------------------------
template <bool OUT16>
__global__ void gemm_nt_kernel(const half_t* __restrict__ A,
                               const half_t* __restrict__ B,
                               const float* __restrict__ Cin,
                               void* __restrict__ Cout, int M, int N, int K,
                               int lda, int ldb, int ldc, float alpha) {
  __shared__ half_t a_lds[64 * 32];    // [m][k]
  __shared__ half_t b_lds[128 * 32];   // [n][k]
  const int tid  = threadIdx.x;
  const int lane = tid & 31;
  const int wave = tid >> 5;
  const int mg = blockIdx.y * 64;
  const int ng = blockIdx.x * 128;
  const int rl = (wave & 3) * 16;   // wave row inside tile
  const int cl = (wave >> 2) * 64;  // wave col inside tile
  const int l16 = lane & 15;
  const int hi  = (lane < 16) ? 0 : 1;
  const int ab  = hi ? 16 : 0;
  const int bkb = hi ? 16 : 0;

  v8f acc[4] = {};
  for (int k0 = 0; k0 < K; k0 += 32) {
    __syncthreads();  // previous iteration's LDS reads complete
    {  // stage A tile: 256 chunks of 8 halfs, one per thread
      const int m = tid >> 2, cc = tid & 3;
      g2l_b128(A + (size_t)(mg + m) * lda + k0 + cc * 8, &a_lds[m * 32 + cc * 8]);
    }
    for (int c = tid; c < 512; c += 256) {  // stage B tile: 512 chunks
      const int n = c >> 2, cc = c & 3;
      g2l_b128(B + (size_t)(ng + n) * ldb + k0 + cc * 8, &b_lds[n * 32 + cc * 8]);
    }
    if (k0 + 32 < K)  // hint next K tile toward L2 (global_prefetch_b8)
      __builtin_prefetch(B + (size_t)(ng + (tid >> 1)) * ldb + k0 + 32, 0, 0);
    async_wait();
    __syncthreads();
    // compute: 1 A frag + 4 B frags -> 4 WMMAs per wave (all b128 LDS reads)
    v16h a;
#pragma unroll
    for (int j = 0; j < 16; ++j)
      a[j] = a_lds[(rl + l16) * 32 + ab + a_k_of(j)];
#pragma unroll
    for (int t = 0; t < 4; ++t) {
      v16h bf;
#pragma unroll
      for (int j = 0; j < 16; ++j)
        bf[j] = b_lds[(cl + t * 16 + l16) * 32 + bkb + j];
      acc[t] = wmma16(a, bf, acc[t]);
    }
  }
#pragma unroll
  for (int t = 0; t < 4; ++t) {
    const int n = ng + cl + t * 16 + l16;
#pragma unroll
    for (int r = 0; r < 8; ++r) {
      const int m = mg + rl + r + hi * 8;
      float v = acc[t][r];
      if (Cin) v += Cin[(size_t)m * ldc + n];
      v *= alpha;
      if (OUT16)
        ((half_t*)Cout)[(size_t)m * ldc + n] = (half_t)v;
      else
        ((float*)Cout)[(size_t)m * ldc + n] = v;
    }
  }
}

// f16 transpose: Y[C][R] = X[R][C]^T, LDS-tiled, coalesced on both sides.
// grid (C/32, R/32), block (32, 8).
__global__ void transpose_f16_kernel(const half_t* __restrict__ X,
                                     half_t* __restrict__ Y, int R, int C) {
  __shared__ half_t tile[32][33];
  const int rb = blockIdx.y * 32, cb = blockIdx.x * 32;
  const int tx = threadIdx.x, ty = threadIdx.y;
#pragma unroll
  for (int i = 0; i < 32; i += 8)
    tile[ty + i][tx] = X[(size_t)(rb + ty + i) * C + cb + tx];
  __syncthreads();
#pragma unroll
  for (int i = 0; i < 32; i += 8)
    Y[(size_t)(cb + ty + i) * R + rb + tx] = tile[tx][ty + i];
}

// ------------------------- elementwise kernels ----------------------------
__global__ void cvt_f16_kernel(const float* __restrict__ X,
                               half_t* __restrict__ Y, size_t n) {
  size_t i = (size_t)blockIdx.x * 256 + threadIdx.x;
  if (i < n) Y[i] = (half_t)X[i];
}

__global__ void rmsnorm_f16_kernel(const float* __restrict__ X,
                                   half_t* __restrict__ Y, int C) {
  __shared__ float red[256];
  const size_t row = blockIdx.x;
  const float* x = X + row * C;
  float s = 0.f;
  for (int c = threadIdx.x; c < C; c += 256) { float v = x[c]; s += v * v; }
  red[threadIdx.x] = s;
  __syncthreads();
  for (int o = 128; o > 0; o >>= 1) {
    if ((int)threadIdx.x < o) red[threadIdx.x] += red[threadIdx.x + o];
    __syncthreads();
  }
  const float inv = rsqrtf(red[0] / (float)C + RMS_EPS);
  for (int c = threadIdx.x; c < C; c += 256) Y[row * C + c] = (half_t)(x[c] * inv);
}

__global__ void aft_rms_kernel(float* __restrict__ QKV) {
  __shared__ float red[256];
  const int m = blockIdx.x >> 1, part = blockIdx.x & 1;
  float* x = QKV + (size_t)m * QKVW + part * Qd;
  float s = 0.f;
  for (int c = threadIdx.x; c < Qd; c += 256) { float v = x[c]; s += v * v; }
  red[threadIdx.x] = s;
  __syncthreads();
  for (int o = 128; o > 0; o >>= 1) {
    if ((int)threadIdx.x < o) red[threadIdx.x] += red[threadIdx.x + o];
    __syncthreads();
  }
  const float inv = rsqrtf(red[0] / (float)Qd + RMS_EPS);
  for (int c = threadIdx.x; c < Qd; c += 256) x[c] *= inv;
}

__global__ void swiglu_uv_kernel(const float* __restrict__ C,
                                 half_t* __restrict__ U,
                                 half_t* __restrict__ SV,
                                 half_t* __restrict__ USV, int width, int ld,
                                 size_t total) {
  size_t i = (size_t)blockIdx.x * 256 + threadIdx.x;
  if (i >= total) return;
  const size_t m = i / width, e = i % width;
  const float u  = C[m * ld + e];
  const float v  = C[m * ld + width + e];
  const float sv = v / (1.f + expf(-v));
  if (U)  U[i]  = (half_t)u;
  if (SV) SV[i] = (half_t)sv;
  USV[i] = (half_t)(u * sv);
}

__global__ void build_parts_kernel(const float* __restrict__ UV,
                                   float* __restrict__ P32,
                                   half_t* __restrict__ P16) {
  __shared__ float red[256];
  const int r = blockIdx.x;
  const float* uv = UV + (size_t)r * E;
  float s = 0.f;
  for (int c = threadIdx.x; c < E; c += 256) { float v = uv[c]; s += v * v; }
  red[threadIdx.x] = s;
  __syncthreads();
  for (int o = 128; o > 0; o >>= 1) {
    if ((int)threadIdx.x < o) red[threadIdx.x] += red[threadIdx.x + o];
    __syncthreads();
  }
  const float inv = rsqrtf(red[0] / (float)E + RMS_EPS);
  for (int c = threadIdx.x; c < E; c += 256) {
    const float w  = uv[c];
    const float th = tanhf(w) * 3.14f;
    float vals[12];
    vals[0] = 1.0f;
    vals[1] = w * inv;
#pragma unroll
    for (int mm = 1; mm <= 5; ++mm) {
      vals[1 + mm] = sinf(th * (float)mm);
      vals[6 + mm] = cosf(th * (float)mm);
    }
#pragma unroll
    for (int blk = 0; blk < 12; ++blk) {
      const size_t o = ((size_t)blk * E + r) * E + c;
      P32[o] = vals[blk];
      P16[o] = (half_t)vals[blk];
    }
  }
}

__global__ void aft_scan_kernel(const float* __restrict__ QKV,
                                half_t* __restrict__ O) {
  const int i = blockIdx.x * 256 + threadIdx.x;  // 0..8191
  const int b = i >> 11, f = i & 2047;
  float num = 0.f, den = 0.f;
  for (int t = 0; t < Tt; ++t) {
    const float* row = QKV + (size_t)(b * Tt + t) * QKVW;
    __builtin_prefetch(row + QKVW, 0, 0);
    const float q = row[f], k = row[Qd + f], v = row[2 * Qd + f];
    const float w = expf(k);
    num += w * v;
    den += w;
    const float sig = 1.f / (1.f + expf(-q));
    O[(size_t)(b * Tt + t) * Qd + f] = (half_t)(sig * num / (den + 1e-6f));
  }
}

__global__ void rope_rms_pack_kernel(const float* __restrict__ QKV,
                                     const float* __restrict__ Cs,
                                     const float* __restrict__ Sn,
                                     half_t* __restrict__ Qo,
                                     half_t* __restrict__ Ko,
                                     half_t* __restrict__ Vo) {
  const int idx = blockIdx.x;  // (b*T + t)*H + h
  const int h  = idx % Hh;
  const int bt = idx / Hh;
  const int t  = bt % Tt;
  const int b  = bt / Tt;
  const int d  = threadIdx.x;  // 0..127
  const float* row = QKV + (size_t)bt * QKVW + (size_t)h * (3 * Dh);
  float qv, kv;
  if (d < 64) {
    const float c = Cs[t * 64 + d], s = Sn[t * 64 + d];
    qv = row[d] * c + row[64 + d] * s;
    kv = row[Dh + d] * c + row[Dh + 64 + d] * s;
  } else {
    const int dd = d - 64;
    const float c = Cs[t * 64 + dd], s = Sn[t * 64 + dd];
    qv = -row[dd] * s + row[d] * c;
    kv = -row[Dh + dd] * s + row[Dh + d] * c;
  }
  const float vv = row[2 * Dh + d];
  __shared__ float rq[128], rk[128];
  rq[d] = qv * qv;
  rk[d] = kv * kv;
  __syncthreads();
  for (int o = 64; o > 0; o >>= 1) {
    if (d < o) { rq[d] += rq[d + o]; rk[d] += rk[d + o]; }
    __syncthreads();
  }
  const float qi = rsqrtf(rq[0] / (float)Dh + RMS_EPS) * 0.08838834764831845f;
  const float ki = rsqrtf(rk[0] / (float)Dh + RMS_EPS);
  const size_t o = ((size_t)((b * Hh + h) * Tt + t)) * Dh + d;
  Qo[o] = (half_t)(qv * qi);
  Ko[o] = (half_t)(kv * ki);
  Vo[o] = (half_t)vv;
}

// Flash attention: grid (B*H, T/64), block 128 = 4 waves, wave = 16 q rows.
// Q (64x128) async-staged once, K (32x128) async-staged per step,
// V transposed during staging into [d][k] for contiguous P@V fragments.
__global__ void flash_attn_kernel(const half_t* __restrict__ Qm,
                                  const half_t* __restrict__ Km,
                                  const half_t* __restrict__ Vm,
                                  half_t* __restrict__ O) {
  __shared__ half_t q_lds[64 * Dh];      // 16KB [q][d]
  __shared__ half_t k_lds[32 * Dh];      //  8KB [k][d]
  __shared__ half_t vT_lds[Dh * 32];     //  8KB [d][k]
  __shared__ half_t p_lds[4][16 * 32];   //  4KB
  const int tid  = threadIdx.x;
  const int lane = tid & 31;
  const int wave = tid >> 5;
  const int bh = blockIdx.x;
  const int b = bh >> 4, h = bh & 15;
  const int qbase = blockIdx.y * 64;
  const int q0 = qbase + wave * 16;
  const half_t* Qp = Qm + (size_t)bh * Tt * Dh;
  const half_t* Kp = Km + (size_t)bh * Tt * Dh;
  const half_t* Vp = Vm + (size_t)bh * Tt * Dh;
  const int l16 = lane & 15;
  const int hi  = (lane < 16) ? 0 : 1;
  const int ab  = hi ? 16 : 0;
  const int bkb = hi ? 16 : 0;

  // stage the block's 64x128 Q tile (1024 chunks of 8 halfs)
  for (int c = tid; c < 64 * (Dh / 8); c += 128) {
    const int r = c >> 4, cc = c & 15;
    g2l_b128(Qp + (size_t)(qbase + r) * Dh + cc * 8, &q_lds[r * Dh + cc * 8]);
  }

  v8f acc[8] = {};
  float mi[8], li[8];
#pragma unroll
  for (int r = 0; r < 8; ++r) { mi[r] = -1e30f; li[r] = 0.f; }

  const int kend = qbase + 64;  // causal bound, uniform per block
  for (int kt = 0; kt < kend; kt += 32) {
    __syncthreads();  // previous step's k/v reads complete
    for (int c = tid; c < 32 * (Dh / 8); c += 128) {  // 512 chunks
      const int r = c >> 4, cc = c & 15;
      g2l_b128(Kp + (size_t)(kt + r) * Dh + cc * 8, &k_lds[r * Dh + cc * 8]);
      // V: global-contiguous b128 read, transpose-scatter into [d][k]
      v8h vv = *(const v8h*)(Vp + (size_t)(kt + r) * Dh + cc * 8);
#pragma unroll
      for (int j = 0; j < 8; ++j) vT_lds[(cc * 8 + j) * 32 + r] = vv[j];
      if (kt + 32 < kend) {
        __builtin_prefetch(Kp + (size_t)(kt + 32 + r) * Dh + cc * 8, 0, 0);
        __builtin_prefetch(Vp + (size_t)(kt + 32 + r) * Dh + cc * 8, 0, 0);
      }
    }
    async_wait();
    __syncthreads();
    if (kt > q0 + 15) continue;  // this wave fully masked for this tile

    const bool need_mask = (kt + 31 > q0);
    // S = Q @ K^T : 16x32 as two 16x16 frags
    v8f s[2] = {};
#pragma unroll
    for (int kc = 0; kc < 4; ++kc) {
      v16h qa;
#pragma unroll
      for (int j = 0; j < 16; ++j)
        qa[j] = q_lds[(wave * 16 + l16) * Dh + kc * 32 + ab + a_k_of(j)];
#pragma unroll
      for (int ng = 0; ng < 2; ++ng) {
        v16h kb;
#pragma unroll
        for (int j = 0; j < 16; ++j)
          kb[j] = k_lds[(ng * 16 + l16) * Dh + kc * 32 + bkb + j];
        s[ng] = wmma16(qa, kb, s[ng]);
      }
    }
    // online softmax per row (row = q0 + r + 8*hi)
#pragma unroll
    for (int r = 0; r < 8; ++r) {
      const int qrow = q0 + r + hi * 8;
      if (need_mask) {
#pragma unroll
        for (int ng = 0; ng < 2; ++ng)
          if (kt + ng * 16 + l16 > qrow) s[ng][r] = -1e30f;
      }
      float mx = fmaxf(s[0][r], s[1][r]);
#pragma unroll
      for (int msk = 1; msk < 16; msk <<= 1)
        mx = fmaxf(mx, __shfl_xor(mx, msk, 32));
      const float mnew = fmaxf(mi[r], mx);
      const float p0 = expf(s[0][r] - mnew);
      const float p1 = expf(s[1][r] - mnew);
      float rs = p0 + p1;
#pragma unroll
      for (int msk = 1; msk < 16; msk <<= 1) rs += __shfl_xor(rs, msk, 32);
      const float corr = expf(mi[r] - mnew);
      li[r] = li[r] * corr + rs;
      mi[r] = mnew;
#pragma unroll
      for (int t = 0; t < 8; ++t) acc[t][r] = acc[t][r] * corr;
      const int mrow = r + hi * 8;
      p_lds[wave][mrow * 32 + l16]      = (half_t)p0;
      p_lds[wave][mrow * 32 + 16 + l16] = (half_t)p1;
    }
    // intra-wave C-layout -> A-layout bounce; LDS is in-order per wave
    __builtin_amdgcn_wave_barrier();
    v16h pa;
#pragma unroll
    for (int j = 0; j < 16; ++j)
      pa[j] = p_lds[wave][l16 * 32 + ab + a_k_of(j)];
    // acc += P @ V  (D = 128 -> 8 frags, contiguous reads from vT)
#pragma unroll
    for (int dc = 0; dc < 8; ++dc) {
      v16h vb;
#pragma unroll
      for (int j = 0; j < 16; ++j)
        vb[j] = vT_lds[(dc * 16 + l16) * 32 + bkb + j];
      acc[dc] = wmma16(pa, vb, acc[dc]);
    }
  }
#pragma unroll
  for (int r = 0; r < 8; ++r) {
    const int qrow = q0 + r + hi * 8;
    const float inv = 1.0f / li[r];
    const size_t base = ((size_t)(b * Tt + qrow)) * Qd + (size_t)h * Dh;
#pragma unroll
    for (int dc = 0; dc < 8; ++dc)
      O[base + dc * 16 + l16] = (half_t)(acc[dc][r] * inv);
  }
}

// --------------------------------------------------------------------------
template <bool O16>
static inline void gemm(hipStream_t s, const half_t* A, const half_t* B,
                        const float* Cin, void* Cout, int M, int N, int K,
                        int lda, int ldb, int ldc, float alpha) {
  dim3 g(N / 128, M / 64);
  gemm_nt_kernel<O16><<<g, 256, 0, s>>>(A, B, Cin, Cout, M, N, K, lda, ldb,
                                        ldc, alpha);
}
static inline void transpose16(hipStream_t s, const half_t* X, half_t* Y,
                               int R, int C) {
  transpose_f16_kernel<<<dim3(C / 32, R / 32), dim3(32, 8), 0, s>>>(X, Y, R, C);
}

extern "C" void kernel_launch(void* const* d_in, const int* in_sizes, int n_in,
                              void* d_out, int out_size, void* d_ws,
                              size_t ws_size, hipStream_t stream) {
  (void)in_sizes; (void)n_in; (void)out_size; (void)ws_size;
  const float* x_in      = (const float*)d_in[0];
  const float* cosp      = (const float*)d_in[1];
  const float* sinp      = (const float*)d_in[2];
  const float* swiglu_w  = (const float*)d_in[3];
  const float* ret_out_w = (const float*)d_in[4];
  const float* sw2_out_w = (const float*)d_in[5];

  char* ws = (char*)d_ws;
  size_t off = 0;
  auto alloc = [&](size_t bytes) -> void* {
    void* p = ws + off;
    off = (off + bytes + 255) & ~(size_t)255;
    return p;
  };
  float*  xf   = (float*)alloc((size_t)Mr * E * 4);
  half_t* sw16 = (half_t*)alloc((size_t)Qd * E * 2);
  half_t* ro16 = (half_t*)alloc((size_t)E * E * 2);
  half_t* s216 = (half_t*)alloc((size_t)E * Qd * 2);
  half_t* W16  = (half_t*)alloc((size_t)Pr * E * 2);
  half_t* A16  = (half_t*)alloc((size_t)Pr * E * 2);
  half_t* P16b = (half_t*)alloc((size_t)Pr * E * 2);
  float*  P32b = (float*)alloc((size_t)Pr * E * 4);
  half_t* U16  = (half_t*)alloc((size_t)Mr * E * 2);
  half_t* SV16 = (half_t*)alloc((size_t)Mr * E * 2);
  half_t* USV16= (half_t*)alloc((size_t)Mr * E * 2);
  half_t* UT16 = (half_t*)alloc((size_t)Mr * E * 2);   // u^T   [E][Mr]
  half_t* SVT16= (half_t*)alloc((size_t)Mr * E * 2);   // sv^T  [E][Mr]
  half_t* WoT16= (half_t*)alloc((size_t)E * Qd * 2);   // w_out [n][k]
  float*  C32  = (float*)alloc((size_t)Pr * Qd * 4);   // == Mr x 6144 too
  float*  UV32 = (float*)alloc((size_t)E * E * 4);
  half_t* Q16  = (half_t*)alloc((size_t)Mr * Qd * 2);
  half_t* K16  = (half_t*)alloc((size_t)Mr * Qd * 2);
  half_t* V16  = (half_t*)alloc((size_t)Mr * Qd * 2);
  half_t* AT16 = (half_t*)alloc((size_t)Mr * Qd * 2);

  const float wscale = 1.0f / sqrtf((float)(6 * Qd));

  (void)hipMemcpyAsync(xf, x_in, (size_t)Mr * E * 4, hipMemcpyDeviceToDevice, stream);
  cvt_f16_kernel<<<(Qd * E) / 256, 256, 0, stream>>>(swiglu_w, sw16, (size_t)Qd * E);
  cvt_f16_kernel<<<(E * E) / 256, 256, 0, stream>>>(ret_out_w, ro16, (size_t)E * E);
  cvt_f16_kernel<<<(E * Qd) / 256, 256, 0, stream>>>(sw2_out_w, s216, (size_t)E * Qd);

  for (int layer = 0; layer < 4; ++layer) {
    // ---- retention ----
    rmsnorm_f16_kernel<<<Mr, 256, 0, stream>>>(xf, A16, E);
    gemm<false>(stream, A16, sw16, nullptr, C32, Mr, Qd, E, E, E, Qd, 1.f);
    swiglu_uv_kernel<<<((size_t)Mr * E) / 256, 256, 0, stream>>>(
        C32, U16, SV16, USV16, E, Qd, (size_t)Mr * E);
    // y = x + (u*silu(v)) @ out_w^T
    gemm<false>(stream, USV16, ro16, xf, xf, Mr, E, E, E, E, E, 1.f);
    // uv = u^T @ sv / B   (as NT GEMM on pre-transposed operands)
    transpose16(stream, U16, UT16, Mr, E);
    transpose16(stream, SV16, SVT16, Mr, E);
    gemm<false>(stream, UT16, SVT16, nullptr, UV32, E, E, Mr, Mr, Mr, E, 0.25f);
    build_parts_kernel<<<E, 256, 0, stream>>>(UV32, P32b, P16b);
    gemm<false>(stream, P16b, sw16, nullptr, C32, Pr, Qd, E, E, E, Qd, 1.f);
    swiglu_uv_kernel<<<((size_t)Pr * E) / 256, 256, 0, stream>>>(
        C32, nullptr, nullptr, A16, E, Qd, (size_t)Pr * E);
    // W = (parts + u2*silu(v2) @ out_w^T) * (6*N_QKV)^-1/2  -> f16 weights
    gemm<true>(stream, A16, ro16, P32b, W16, Pr, E, E, E, E, E, wscale);
    // pre-transpose w_out rows [3Q,4Q): [2048][1024] -> [1024][2048]
    transpose16(stream, W16 + (size_t)3 * Qd * E, WoT16, Qd, E);

    // ---- qkv projection ----
    rmsnorm_f16_kernel<<<Mr, 256, 0, stream>>>(xf, A16, E);
    gemm<false>(stream, A16, W16, nullptr, C32, Mr, QKVW, E, E, E, QKVW, 1.f);

    // ---- attention ----
    if (layer == 3) {
      rope_rms_pack_kernel<<<Mr * Hh, 128, 0, stream>>>(C32, cosp, sinp, Q16, K16, V16);
      flash_attn_kernel<<<dim3(Bb * Hh, Tt / 64), 128, 0, stream>>>(Q16, K16, V16, AT16);
    } else {
      aft_rms_kernel<<<2 * Mr, 256, 0, stream>>>(C32);
      aft_scan_kernel<<<(Bb * Qd) / 256, 256, 0, stream>>>(C32, AT16);
    }
    // x += attn @ w_out^T   (WoT16 is [n][k] -> plain NT GEMM)
    gemm<false>(stream, AT16, WoT16, xf, xf, Mr, E, Qd, Qd, Qd, E, 1.f);

    // ---- final swiglu ----
    rmsnorm_f16_kernel<<<Mr, 256, 0, stream>>>(xf, A16, E);
    gemm<false>(stream, A16, W16 + (size_t)4 * Qd * E, nullptr, C32,
                Mr, 2 * Qd, E, E, E, 2 * Qd, 1.f);
    swiglu_uv_kernel<<<((size_t)Mr * Qd) / 256, 256, 0, stream>>>(
        C32, nullptr, nullptr, AT16, Qd, 2 * Qd, (size_t)Mr * Qd);
    gemm<false>(stream, AT16, s216, xf, xf, Mr, E, Qd, Qd, Qd, E, 1.f);
  }

  (void)hipMemcpyAsync(d_out, xf, (size_t)Mr * E * 4, hipMemcpyDeviceToDevice, stream);
}
